// DagGenGRUg4s2v1_21174188769804
// MI455X (gfx1250) — compile-verified, matrix operand order checked by reading, not run on hardware
//
#include <hip/hip_runtime.h>

typedef __attribute__((ext_vector_type(16))) _Float16 v16h;
typedef __attribute__((ext_vector_type(8)))  float    v8f;
typedef __attribute__((ext_vector_type(2)))  float    v2f;

// ---------------- workspace layout (in halves for the f16 region) -------------
static constexpr size_t H_OFF    = 0;
static constexpr size_t H_SZ     = (size_t)17 * 32 * 64 * 64;      // encoder output h, f16
static constexpr size_t SC_SZ    = (size_t)16 * 32 * 64 * 64 * 4;  // scores, head-minor, f16
static constexpr size_t SCI_OFF  = H_OFF + H_SZ;
static constexpr size_t SCU_OFF  = SCI_OFF + SC_SZ;
static constexpr size_t WENC1    = SCU_OFF + SC_SZ;   // swizzled B-frag layout
static constexpr size_t WENC2    = WENC1 + 4096;
static constexpr size_t WIQ_OFF  = WENC2 + 4096;
static constexpr size_t WIK_OFF  = WIQ_OFF + 4096;
static constexpr size_t WUQ_OFF  = WIK_OFF + 4096;
static constexpr size_t WUK_OFF  = WUQ_OFF + 4096;
static constexpr size_t GW0_OFF  = WUK_OFF + 4096;  // gi GRU pack (8192 halves)
static constexpr size_t GW1_OFF  = GW0_OFF + 8192;  // gu GRU pack
static constexpr size_t WS_HALVES = GW1_OFF + 8192;
static constexpr size_t GU_BYTE   = ((WS_HALVES * 2 + 255) / 256) * 256;
static constexpr size_t GU_FLOATS = (size_t)16 * 32 * 64 * 64;

// GRU pack sub-offsets (halves): all K-padded to 32 rows, B-frag swizzled
#define P_WIH0 0
#define P_WHH0 1536
#define P_WIH1 3072
#define P_WHH1 4608
#define P_W1   6144

// ---------------- WMMA wrappers ----------------------------------------------
__device__ __forceinline__ v8f z8() {
  v8f z;
#pragma unroll
  for (int e = 0; e < 8; ++e) z[e] = 0.0f;
  return z;
}

__device__ __forceinline__ v8f wmma_f16(v16h a, v16h b, v8f c) {
  return __builtin_amdgcn_wmma_f32_16x16x32_f16(false, a, false, b, (short)0, c,
                                                false, false);
}

#if defined(__has_builtin)
#if __has_builtin(__builtin_amdgcn_wmma_f32_16x16x4_f32)
#define HAVE_WMMA_F32_K4 1
#endif
#endif

#ifdef HAVE_WMMA_F32_K4
__device__ __forceinline__ v8f wmma_f32k4(v2f a, v2f b, v8f c) {
  return __builtin_amdgcn_wmma_f32_16x16x4_f32(false, a, false, b, (short)0, c,
                                               false, false);
}
#else
__device__ __forceinline__ v8f wmma_f32k4(v2f a, v2f b, v8f c) {
#pragma unroll
  for (int e = 0; e < 8; ++e) c[e] += a.x * b.x + a.y * b.y;
  return c;
}
#endif

// ---------------- fragment loaders (per ISA 7.12.2 layouts) -------------------
// A 16x32 f16 from row-major source, ld halves (even), k0 in {0,32}.
// Lane's 16 halves = [kb..kb+7] and [kb+16..kb+23]: two 128-bit loads.
__device__ __forceinline__ v16h load_a_f16(const _Float16* A, int ld, int r0, int k0) {
  int l = threadIdx.x & 31;
  int m = r0 + (l & 15);
  int kb = k0 + ((l >> 4) << 3);
  const _Float16* p = A + (size_t)m * ld + kb;
  union { v16h v; uint4 q[2]; } r;
  r.q[0] = *(const uint4*)p;
  r.q[1] = *(const uint4*)(p + 16);
  return r.v;
}

// B 32x16 f16 fragment from pre-swizzled storage:
// layout [k_chunk][n_tile][lane 0..31][16 halves] -> 2 x b128 per lane.
__device__ __forceinline__ v16h load_b_sw(const _Float16* B, int ntiles, int kc,
                                          int ntile) {
  int l = threadIdx.x & 31;
  const _Float16* p = B + (((size_t)(kc * ntiles + ntile) * 32 + l) << 4);
  union { v16h v; uint4 q[2]; } r;
  r.q[0] = *(const uint4*)p;
  r.q[1] = *(const uint4*)(p + 8);
  return r.v;
}

// A 16x4 f32 (pairs contiguous -> one 8B load). Also used for B^T (scores).
__device__ __forceinline__ v2f load_a_f32(const float* A, int ld, int r0, int k0) {
  int l = threadIdx.x & 31;
  int m = r0 + (l & 15);
  int k = k0 + ((l >> 4) << 1);
  return *(const v2f*)(A + (size_t)m * ld + k);
}

// B 4x16 f32 from row-major KxN source (row-strided, 2 scalar loads).
__device__ __forceinline__ v2f load_b_f32(const float* B, int ld, int c0, int k0) {
  int l = threadIdx.x & 31;
  int n = c0 + (l & 15);
  int k = k0 + ((l >> 4) << 1);
  v2f r; r.x = B[(size_t)k * ld + n]; r.y = B[(size_t)(k + 1) * ld + n];
  return r;
}

// A 16x4 f32 of M = I + W (identity on the fly), ld = 64.
__device__ __forceinline__ v2f load_a_Mf32(const float* W, int r0, int k0) {
  int l = threadIdx.x & 31;
  int m = r0 + (l & 15);
  int k = k0 + ((l >> 4) << 1);
  v2f r = *(const v2f*)(W + m * 64 + k);
  r.x += (m == k) ? 1.0f : 0.0f;
  r.y += (m == k + 1) ? 1.0f : 0.0f;
  return r;
}

__device__ __forceinline__ void store_acc_f32(float* C, int ld, int r0, int c0, v8f a) {
  int l = threadIdx.x & 31;
#pragma unroll
  for (int e = 0; e < 8; ++e)
    C[(size_t)(r0 + e + ((l >> 4) << 3)) * ld + c0 + (l & 15)] = a[e];
}

__device__ __forceinline__ void store_acc_f16(_Float16* C, int ld, v8f a) {
  int l = threadIdx.x & 31;
#pragma unroll
  for (int e = 0; e < 8; ++e)
    C[(size_t)(e + ((l >> 4) << 3)) * ld + (l & 15)] = (_Float16)a[e];
}

__device__ __forceinline__ float sigmf(float x) { return 1.0f / (1.0f + __expf(-x)); }

// ============================= K0: weight prep ================================
// Converts f32 weights to f16 pre-swizzled B-fragment order:
//   dst[((kc*ntiles + ntile)*32 + lane)*16 + e] = src[k, n]
//   n = ntile*16 + (lane&15)
//   k = kc*32 + 8*(lane>>4) + ((e>>1)&3)*2 + ((e>>1)>>2)*16 + (e&1)
__global__ void prep_kernel(const float* eW1, const float* eW2,
                            const float* Wiq, const float* Wik,
                            const float* Wuq, const float* Wuk,
                            const float* giWih0, const float* giWhh0,
                            const float* giWih1, const float* giWhh1,
                            const float* dgiW1,
                            const float* guWih0, const float* guWhh0,
                            const float* guWih1, const float* guWhh1,
                            const float* dguW1,
                            _Float16* __restrict__ wsh) {
  int tid = threadIdx.x, nthr = blockDim.x;
  auto swz = [&](const float* s, size_t off, int K, int N, int realK) {
    int ntiles = N >> 4;
    int total = K * N;
    for (int i = tid; i < total; i += nthr) {
      int e = i & 15;
      int lane = (i >> 4) & 31;
      int blk = i >> 9;
      int kc = blk / ntiles, ntile = blk % ntiles;
      int n = ntile * 16 + (lane & 15);
      int p = e >> 1;
      int k = kc * 32 + ((lane >> 4) << 3) + ((p & 3) << 1) + ((p >> 2) << 4) +
              (e & 1);
      wsh[off + i] = (k < realK) ? (_Float16)s[k * N + n] : (_Float16)0.0f;
    }
  };
  swz(eW1, WENC1, 64, 64, 64);
  swz(eW2, WENC2, 64, 64, 64);
  swz(Wiq, WIQ_OFF, 64, 64, 64);
  swz(Wik, WIK_OFF, 64, 64, 64);
  swz(Wuq, WUQ_OFF, 64, 64, 64);
  swz(Wuk, WUK_OFF, 64, 64, 64);
  swz(giWih0, GW0_OFF + P_WIH0, 32, 48, 4);
  swz(giWhh0, GW0_OFF + P_WHH0, 32, 48, 16);
  swz(giWih1, GW0_OFF + P_WIH1, 32, 48, 16);
  swz(giWhh1, GW0_OFF + P_WHH1, 32, 48, 16);
  swz(dgiW1,  GW0_OFF + P_W1,   32, 64, 16);
  swz(guWih0, GW1_OFF + P_WIH0, 32, 48, 4);
  swz(guWhh0, GW1_OFF + P_WHH0, 32, 48, 16);
  swz(guWih1, GW1_OFF + P_WIH1, 32, 48, 16);
  swz(guWhh1, GW1_OFF + P_WHH1, 32, 48, 16);
  swz(dguW1,  GW1_OFF + P_W1,   32, 64, 16);
}

// ============================= K1: encoder MLP ================================
// rows = 17*32*64 = 34816; block = 128 (4 waves), 16 rows/wave -> grid 544
__global__ void enc_kernel(const float* __restrict__ x,
                           const float* eW0, const float* eb0,
                           const float* eb1, const float* eb2,
                           _Float16* __restrict__ wsh) {
  __shared__ alignas(16) _Float16 sW1[4096], sW2[4096];
  __shared__ float sW0[128], sb0[64], sb1[64], sb2[64];
  __shared__ alignas(16) _Float16 sAct[4][16 * 64];
  for (int i = threadIdx.x; i < 4096; i += 128) {
    sW1[i] = wsh[WENC1 + i];
    sW2[i] = wsh[WENC2 + i];
  }
  if (threadIdx.x < 128) sW0[threadIdx.x] = eW0[threadIdx.x];
  if (threadIdx.x < 64) {
    sb0[threadIdx.x] = eb0[threadIdx.x];
    sb1[threadIdx.x] = eb1[threadIdx.x];
    sb2[threadIdx.x] = eb2[threadIdx.x];
  }
  __syncthreads();

  int wid = threadIdx.x >> 5, l = threadIdx.x & 31;
  int rbase = blockIdx.x * 64 + wid * 16;
  _Float16* act = sAct[wid];

  // layer 0 (IN_F=2 -> 64) scalar, relu, into f16 slab
  {
    int row = l & 15, ch = (l >> 4) * 32;
    const float* xr = x + (size_t)(rbase + row) * 2;
    __builtin_prefetch(xr, 0, 0);
    float x0 = xr[0], x1 = xr[1];
#pragma unroll
    for (int c = 0; c < 32; ++c) {
      int cc = ch + c;
      float v = x0 * sW0[cc] + x1 * sW0[64 + cc] + sb0[cc];
      act[row * 64 + cc] = (_Float16)fmaxf(v, 0.0f);
    }
  }
  // layer 1: 16x64 @ 64x64 (WMMA) + relu, back into slab
  {
    v16h a0 = load_a_f16(act, 64, 0, 0), a1 = load_a_f16(act, 64, 0, 32);
    v8f acc[4];
#pragma unroll
    for (int nt = 0; nt < 4; ++nt) {
      acc[nt] = wmma_f16(a0, load_b_sw(sW1, 4, 0, nt), z8());
      acc[nt] = wmma_f16(a1, load_b_sw(sW1, 4, 1, nt), acc[nt]);
    }
#pragma unroll
    for (int nt = 0; nt < 4; ++nt)
#pragma unroll
      for (int e = 0; e < 8; ++e) {
        int row = e + ((l >> 4) << 3), col = nt * 16 + (l & 15);
        act[row * 64 + col] = (_Float16)fmaxf(acc[nt][e] + sb1[col], 0.0f);
      }
  }
  // layer 2: no relu, store h (f16) to workspace
  {
    v16h a0 = load_a_f16(act, 64, 0, 0), a1 = load_a_f16(act, 64, 0, 32);
    v8f acc[4];
#pragma unroll
    for (int nt = 0; nt < 4; ++nt) {
      acc[nt] = wmma_f16(a0, load_b_sw(sW2, 4, 0, nt), z8());
      acc[nt] = wmma_f16(a1, load_b_sw(sW2, 4, 1, nt), acc[nt]);
    }
    _Float16* h = wsh + H_OFF;
#pragma unroll
    for (int nt = 0; nt < 4; ++nt)
#pragma unroll
      for (int e = 0; e < 8; ++e) {
        int row = e + ((l >> 4) << 3), col = nt * 16 + (l & 15);
        h[(size_t)(rbase + row) * 64 + col] = (_Float16)(acc[nt][e] + sb2[col]);
      }
  }
}

// =================== K2: fused projections + per-head scores ==================
// one block (256 thr, 8 waves) per (t,b); 512 blocks
__global__ void score_kernel(_Float16* __restrict__ wsh) {
  __shared__ alignas(16) _Float16 sWb[2][4096];
  __shared__ alignas(16) float sProj[2][4096];
  int t = blockIdx.x >> 5, b = blockIdx.x & 31;
  const _Float16* hX = wsh + H_OFF + (size_t)((t + 1) * 32 + b) * 4096;  // x = h[t+1]
  const _Float16* hP = wsh + H_OFF + (size_t)(t * 32 + b) * 4096;        // pre = h[t]
  int wid = threadIdx.x >> 5, l = threadIdx.x & 31;

  for (int s = 0; s < 2; ++s) {
    const _Float16* wq = wsh + (s ? WUQ_OFF : WIQ_OFF);
    const _Float16* wk = wsh + (s ? WUK_OFF : WIK_OFF);
    for (int i = threadIdx.x; i < 4096; i += 256) {
      sWb[0][i] = wq[i];
      sWb[1][i] = wk[i];
    }
    __syncthreads();
    // phase 1: Q = X@Wq, K = P@Wk (f16 WMMA, K=64), f32 result to LDS
    for (int j = wid; j < 32; j += 8) {
      int q = j >> 4, tt = j & 15;
      int r0 = (tt >> 2) << 4, c0 = (tt & 3) << 4;
      const _Float16* src = q ? hP : hX;
      v8f acc = z8();
      acc = wmma_f16(load_a_f16(src, 64, r0, 0),  load_b_sw(sWb[q], 4, 0, tt & 3), acc);
      acc = wmma_f16(load_a_f16(src, 64, r0, 32), load_b_sw(sWb[q], 4, 1, tt & 3), acc);
      store_acc_f32(sProj[q], 64, r0, c0, acc);
    }
    __syncthreads();
    // phase 2: per-head score = Q_h @ K_h^T (f32 WMMA, K=16) -> f16 head-minor
    _Float16* sc = wsh + (s ? SCU_OFF : SCI_OFF) +
                   (size_t)t * 524288 + (size_t)b * 16384;
    for (int j = wid; j < 64; j += 8) {
      int head = j >> 4, tt = j & 15;
      int r0 = (tt >> 2) << 4, c0 = (tt & 3) << 4;
      v8f acc = z8();
#pragma unroll
      for (int k = 0; k < 16; k += 4)
        acc = wmma_f32k4(load_a_f32(sProj[0], 64, r0, head * 16 + k),
                         load_a_f32(sProj[1], 64, c0, head * 16 + k), acc);
#pragma unroll
      for (int e = 0; e < 8; ++e) {
        int n = r0 + e + ((l >> 4) << 3), m = c0 + (l & 15);
        sc[(size_t)(n * 64 + m) * 4 + head] = (_Float16)acc[e];
      }
    }
    __syncthreads();
  }
}

// ================ K3: fused 2-layer GRU + dg head (gi & gu) ===================
// 131072 rows; block 128 (4 waves, 16 rows each); grid (2048, 2)
__global__ void gru_kernel(_Float16* __restrict__ wsh,
                           const float* bih0i, const float* bhh0i,
                           const float* bih1i, const float* bhh1i,
                           const float* b1i, const float* w2i, const float* b2i,
                           const float* bih0u, const float* bhh0u,
                           const float* bih1u, const float* bhh1u,
                           const float* b1u, const float* w2u, const float* b2u,
                           float* __restrict__ out, float* __restrict__ guws) {
  int v = blockIdx.y;
  __shared__ alignas(16) _Float16 sGW[8192];
  __shared__ float sbih0[48], sbhh0[48], sbih1[48], sbhh1[48], sb1[64], sw2[64];
  __shared__ float sb2;
  __shared__ alignas(16) _Float16 slab[4][3][16 * 32];

  const _Float16* gw = wsh + (v ? GW1_OFF : GW0_OFF);
  for (int i = threadIdx.x; i < 8192; i += 128) sGW[i] = gw[i];
  if (threadIdx.x < 48) {
    int i = threadIdx.x;
    sbih0[i] = (v ? bih0u : bih0i)[i];
    sbhh0[i] = (v ? bhh0u : bhh0i)[i];
    sbih1[i] = (v ? bih1u : bih1i)[i];
    sbhh1[i] = (v ? bhh1u : bhh1i)[i];
  }
  if (threadIdx.x < 64) {
    sb1[threadIdx.x] = (v ? b1u : b1i)[threadIdx.x];
    sw2[threadIdx.x] = (v ? w2u : w2i)[threadIdx.x];
  }
  if (threadIdx.x == 0) sb2 = (v ? b2u : b2i)[0];
  for (int i = threadIdx.x; i < 4 * 3 * 512; i += 128)
    ((_Float16*)slab)[i] = (_Float16)0.0f;
  __syncthreads();

  // hoist all loop-invariant weight fragments into registers (wave32 VGPR budget)
  v16h fWih0[3], fWhh0[3], fWih1[3], fWhh1[3], fW1[4];
#pragma unroll
  for (int nt = 0; nt < 3; ++nt) {
    fWih0[nt] = load_b_sw(sGW + P_WIH0, 3, 0, nt);
    fWhh0[nt] = load_b_sw(sGW + P_WHH0, 3, 0, nt);
    fWih1[nt] = load_b_sw(sGW + P_WIH1, 3, 0, nt);
    fWhh1[nt] = load_b_sw(sGW + P_WHH1, 3, 0, nt);
  }
#pragma unroll
  for (int nt = 0; nt < 4; ++nt) fW1[nt] = load_b_sw(sGW + P_W1, 4, 0, nt);

  int wid = threadIdx.x >> 5, l = threadIdx.x & 31, cb = l & 15;
  _Float16* xs  = slab[wid][0];
  _Float16* h0s = slab[wid][1];
  _Float16* h1s = slab[wid][2];
  int jbase = (blockIdx.x * 4 + wid) * 16;
  const _Float16* sc = wsh + (v ? SCU_OFF : SCI_OFF);

  v8f h0 = z8(), h1 = z8();
  for (int t = 0; t < 16; ++t) {
    if (l < 16) {  // 4 head features per row, contiguous f16x4
      const _Float16* xr = sc + ((size_t)t * 131072 + jbase + l) * 4;
      *(uint2*)(xs + l * 32) = *(const uint2*)xr;
    }
    v16h ax = load_a_f16(xs, 32, 0, 0);
    store_acc_f16(h0s, 32, h0);              // h0(prev) -> A fragment
    v16h ah0 = load_a_f16(h0s, 32, 0, 0);
    v8f I0[3], H0g[3];
#pragma unroll
    for (int nt = 0; nt < 3; ++nt) {
      I0[nt]  = wmma_f16(ax,  fWih0[nt], z8());
      H0g[nt] = wmma_f16(ah0, fWhh0[nt], z8());
    }
#pragma unroll
    for (int e = 0; e < 8; ++e) {
      float r = sigmf(I0[0][e] + sbih0[cb] + H0g[0][e] + sbhh0[cb]);
      float zz = sigmf(I0[1][e] + sbih0[16 + cb] + H0g[1][e] + sbhh0[16 + cb]);
      float nn = tanhf(I0[2][e] + sbih0[32 + cb] +
                       r * (H0g[2][e] + sbhh0[32 + cb]));
      h0[e] = (1.0f - zz) * nn + zz * h0[e];
    }
    // layer 1: input = new h0
    store_acc_f16(h0s, 32, h0);
    v16h ax1 = load_a_f16(h0s, 32, 0, 0);
    store_acc_f16(h1s, 32, h1);
    v16h ah1 = load_a_f16(h1s, 32, 0, 0);
    v8f I1[3], H1g[3];
#pragma unroll
    for (int nt = 0; nt < 3; ++nt) {
      I1[nt]  = wmma_f16(ax1, fWih1[nt], z8());
      H1g[nt] = wmma_f16(ah1, fWhh1[nt], z8());
    }
#pragma unroll
    for (int e = 0; e < 8; ++e) {
      float r = sigmf(I1[0][e] + sbih1[cb] + H1g[0][e] + sbhh1[cb]);
      float zz = sigmf(I1[1][e] + sbih1[16 + cb] + H1g[1][e] + sbhh1[16 + cb]);
      float nn = tanhf(I1[2][e] + sbih1[32 + cb] +
                       r * (H1g[2][e] + sbhh1[32 + cb]));
      h1[e] = (1.0f - zz) * nn + zz * h1[e];
    }
    // dg head: relu(h1 @ W1 + b1) . w2 + b2 -> sigmoid
    store_acc_f16(h1s, 32, h1);
    v16h ad = load_a_f16(h1s, 32, 0, 0);
    float partial[8];
#pragma unroll
    for (int e = 0; e < 8; ++e) partial[e] = 0.0f;
#pragma unroll
    for (int nt = 0; nt < 4; ++nt) {
      v8f F = wmma_f16(ad, fW1[nt], z8());
      int col = nt * 16 + cb;
#pragma unroll
      for (int e = 0; e < 8; ++e)
        partial[e] += fmaxf(F[e] + sb1[col], 0.0f) * sw2[col];
    }
#pragma unroll
    for (int e = 0; e < 8; ++e) {
      float vs = partial[e];
#pragma unroll
      for (int msk = 1; msk < 16; msk <<= 1) vs += __shfl_xor(vs, msk);
      if (cb == e) {
        int row = e + ((l >> 4) << 3);
        int j = jbase + row;
        float g = sigmf(vs + sb2);
        int bb = j >> 12, nm = j & 4095;
        if (v) {
          int n = nm >> 6, m = nm & 63;
          guws[(size_t)(t * 32 + bb) * 4096 + nm] = (n == m) ? 0.0f : g;
        } else {
          out[((size_t)(t * 32 + bb) * 2) * 4096 + nm] = g;
        }
      }
    }
  }
}

// ============== K4: DAGMA — Gauss-Jordan init + Newton-Schulz WMMA ============
// one block (256 thr, 8 waves) per (t,b) matrix; dynamic LDS = 64KB
__global__ void dagma_kernel(const float* __restrict__ guws,
                             float* __restrict__ out) {
  extern __shared__ float sm[];
  float* Wl = sm;           // W (>=0)
  float* Xa = sm + 4096;    // inverse ping
  float* Xb = sm + 8192;    // inverse pong (also GJ factor scratch)
  float* Tt = sm + 12288;   // temp / GJ working copy of M
  __shared__ float spiv;

  int tb = blockIdx.x;
  int tid = threadIdx.x, wid = tid >> 5;
  const float* src = guws + (size_t)tb * 4096;
  for (int i = tid; i < 4096; i += 256) {
    float w = src[i];
    float ii = ((i >> 6) == (i & 63)) ? 1.0f : 0.0f;
    Wl[i] = w;
    Tt[i] = w + ii;   // M = I + |W|, W >= 0
    Xa[i] = ii;
  }
  __syncthreads();

  // exact inverse once (Gauss-Jordan; M is diag-dominant-ish, no pivoting)
  float* fac = Xb;
  for (int p = 0; p < 64; ++p) {
    if (tid == 0) spiv = 1.0f / Tt[p * 64 + p];
    __syncthreads();
    float rp = spiv;
    if (tid < 128) {
      if (tid < 64) Tt[p * 64 + tid] *= rp;
      else          Xa[p * 64 + (tid - 64)] *= rp;
    }
    __syncthreads();
    if (tid < 64) fac[tid] = (tid == p) ? 0.0f : Tt[tid * 64 + p];
    __syncthreads();
    for (int idx = tid; idx < 8192; idx += 256) {
      int r = idx >> 7, c = idx & 127;
      float f = fac[r];
      if (c < 64) Tt[r * 64 + c] -= f * Tt[p * 64 + c];
      else { int cc = c - 64; Xa[r * 64 + cc] -= f * Xa[p * 64 + cc]; }
    }
    __syncthreads();
  }

  float* Xc = Xa;
  float* Xn = Xb;
  for (int step = 0; step < 150; ++step) {
    if (step > 0) {
      // two Newton-Schulz refinements: X <- X (2I - M X), M = I + W(current)
#pragma unroll 1
      for (int it = 0; it < 2; ++it) {
        for (int tt2 = wid; tt2 < 16; tt2 += 8) {
          int r0 = (tt2 >> 2) << 4, c0 = (tt2 & 3) << 4;
          v8f acc = z8();
#pragma unroll
          for (int k = 0; k < 64; k += 4)
            acc = wmma_f32k4(load_a_Mf32(Wl, r0, k),
                             load_b_f32(Xc, 64, c0, k), acc);
          store_acc_f32(Tt, 64, r0, c0, acc);
        }
        __syncthreads();
        for (int i = tid; i < 4096; i += 256)
          Tt[i] = (((i >> 6) == (i & 63)) ? 2.0f : 0.0f) - Tt[i];
        __syncthreads();
        for (int tt2 = wid; tt2 < 16; tt2 += 8) {
          int r0 = (tt2 >> 2) << 4, c0 = (tt2 & 3) << 4;
          v8f acc = z8();
#pragma unroll
          for (int k = 0; k < 64; k += 4)
            acc = wmma_f32k4(load_a_f32(Xc, 64, r0, k),
                             load_b_f32(Tt, 64, c0, k), acc);
          store_acc_f32(Xn, 64, r0, c0, acc);
        }
        __syncthreads();
        float* tmp = Xc; Xc = Xn; Xn = tmp;
      }
    }
    // W <- max(W + lr * invM .* sign(W), 0)   (sign(W) in {0,1} since W>=0)
    for (int i = tid; i < 4096; i += 256) {
      float w = Wl[i];
      Wl[i] = (w > 0.0f) ? fmaxf(w + 0.01f * Xc[i], 0.0f) : 0.0f;
    }
    __syncthreads();
  }
  float* dst = out + ((size_t)tb * 2 + 1) * 4096;
  for (int i = tid; i < 4096; i += 256) dst[i] = Wl[i];
}

// ============================== host launcher =================================
extern "C" void kernel_launch(void* const* d_in, const int* in_sizes, int n_in,
                              void* d_out, int out_size, void* d_ws, size_t ws_size,
                              hipStream_t stream) {
  (void)in_sizes; (void)n_in; (void)out_size;
  const float* x      = (const float*)d_in[0];
  const float* eW0    = (const float*)d_in[1];
  const float* eb0    = (const float*)d_in[2];
  const float* eW1    = (const float*)d_in[3];
  const float* eb1    = (const float*)d_in[4];
  const float* eW2    = (const float*)d_in[5];
  const float* eb2    = (const float*)d_in[6];
  const float* Wiq    = (const float*)d_in[7];
  const float* Wik    = (const float*)d_in[8];
  const float* Wuq    = (const float*)d_in[9];
  const float* Wuk    = (const float*)d_in[10];
  const float* giWih0 = (const float*)d_in[11];
  const float* giWhh0 = (const float*)d_in[12];
  const float* gibih0 = (const float*)d_in[13];
  const float* gibhh0 = (const float*)d_in[14];
  const float* giWih1 = (const float*)d_in[15];
  const float* giWhh1 = (const float*)d_in[16];
  const float* gibih1 = (const float*)d_in[17];
  const float* gibhh1 = (const float*)d_in[18];
  const float* guWih0 = (const float*)d_in[19];
  const float* guWhh0 = (const float*)d_in[20];
  const float* gubih0 = (const float*)d_in[21];
  const float* gubhh0 = (const float*)d_in[22];
  const float* guWih1 = (const float*)d_in[23];
  const float* guWhh1 = (const float*)d_in[24];
  const float* gubih1 = (const float*)d_in[25];
  const float* gubhh1 = (const float*)d_in[26];
  const float* dgiW1  = (const float*)d_in[27];
  const float* dgib1  = (const float*)d_in[28];
  const float* dgiW2  = (const float*)d_in[29];
  const float* dgib2  = (const float*)d_in[30];
  const float* dguW1  = (const float*)d_in[31];
  const float* dgub1  = (const float*)d_in[32];
  const float* dguW2  = (const float*)d_in[33];
  const float* dgub2  = (const float*)d_in[34];

  if (ws_size < GU_BYTE + GU_FLOATS * sizeof(float)) return;  // need ~47MB
  _Float16* wsh = (_Float16*)d_ws;
  float* guws = (float*)((char*)d_ws + GU_BYTE);
  float* out = (float*)d_out;

  prep_kernel<<<1, 256, 0, stream>>>(eW1, eW2, Wiq, Wik, Wuq, Wuk,
                                     giWih0, giWhh0, giWih1, giWhh1, dgiW1,
                                     guWih0, guWhh0, guWih1, guWhh1, dguW1, wsh);
  enc_kernel<<<544, 128, 0, stream>>>(x, eW0, eb0, eb1, eb2, wsh);
  score_kernel<<<512, 256, 0, stream>>>(wsh);
  gru_kernel<<<dim3(2048, 2), 128, 0, stream>>>(
      wsh, gibih0, gibhh0, gibih1, gibhh1, dgib1, dgiW2, dgib2,
      gubih0, gubhh0, gubih1, gubhh1, dgub1, dguW2, dgub2, out, guws);
  dagma_kernel<<<512, 256, 16384 * sizeof(float), stream>>>(guws, out);
}
